// CLIP_26869315404321
// MI455X (gfx1250) — matrix operand order repaired and checked by reading, not compiled, not run on hardware
//
#include <hip/hip_runtime.h>
#include <hip/hip_bf16.h>

typedef __bf16 bf16_t;
typedef __attribute__((ext_vector_type(16))) __bf16 v16bf;
typedef __attribute__((ext_vector_type(8)))  float  v8f;

#define NROW 4096
#define DDIM 2048
#define PDIM 512
#define LN_EPS 1e-5f
#define MASK_VAL 1000.0f

struct __align__(8) bf16x4 { bf16_t v[4]; };

// ---- WMMA fragment loaders (CDNA5 16-bit layouts, wave32) ----
// A (16x32, MxK): lane m = L%16, h = L/16; elems 0..7 -> k = 8h+e, elems 8..15 -> k = 16+8h+(e-8)
__device__ inline v16bf load_frag_a(const bf16_t* base, int ld, int lane) {
    const int m = lane & 15, h = (lane >> 4) & 1;
    const bf16_t* p = base + m * ld + 8 * h;
    v16bf a;
#pragma unroll
    for (int e = 0; e < 8; ++e) a[e] = p[e];
#pragma unroll
    for (int e = 0; e < 8; ++e) a[8 + e] = p[16 + e];
    return a;
}
// B (32x16, KxN) from row-major [n][k] storage: lane n = L%16, h = L/16; elems 0..7 -> k = 8h+e, elems 8..15 -> k = 16+8h+(e-8)
__device__ inline v16bf load_frag_b(const bf16_t* base, int ld, int lane) {
    const int n = lane & 15, h = (lane >> 4) & 1;
    const bf16_t* p = base + n * ld + 8 * h;
    v16bf b;
#pragma unroll
    for (int e = 0; e < 8; ++e) b[e] = p[e];
#pragma unroll
    for (int e = 0; e < 8; ++e) b[8 + e] = p[16 + e];
    return b;
}

__device__ inline v8f wmma_bf16(v16bf a, v16bf b, v8f c) {
    return __builtin_amdgcn_wmma_f32_16x16x32_bf16(false, a, false, b, (short)0, c, false, false);
}

// Async HBM -> LDS copy, 16 bytes per lane, tracked with ASYNCcnt (gfx1250 TDM-adjacent path).
__device__ inline void async_copy_b128(unsigned lds_byte_addr, const void* gaddr) {
    asm volatile("global_load_async_to_lds_b128 %0, %1, off"
                 :: "v"(lds_byte_addr), "v"((unsigned long long)(uintptr_t)gaddr)
                 : "memory");
}
__device__ inline void wait_async_all() {
    asm volatile("s_wait_asynccnt 0x0" ::: "memory");
}

// ======================= Kernel 1: projection + LayerNorm =======================
// Block: 256 thr (8 waves). Tile: 32 rows x 512 cols (full P -> LN local to block).
__global__ __launch_bounds__(256) void proj_ln_kernel(
    const float* __restrict__ X, const float* __restrict__ W,
    const float* __restrict__ bias, const float* __restrict__ g,
    const float* __restrict__ bvec, bf16_t* __restrict__ out)
{
    __shared__ bf16_t Abf[32][32];        // X tile  [m][k]
    __shared__ bf16_t BbfT[PDIM][32];     // W tile transposed [n][k]
    __shared__ float  red[32][4][2];      // per-row partial (sum, sumsq) per N-group
    __shared__ float  muArr[32], rsArr[32];

    const int tid  = threadIdx.x;
    const int wave = tid >> 5, lane = tid & 31;
    const int mt = wave & 1, ng = wave >> 1;
    const int half = (lane >> 4) & 1, l15 = lane & 31 & 15;
    const int m0 = blockIdx.x * 32;

    v8f acc[8] = {};

    for (int k0 = 0; k0 < DDIM; k0 += 32) {
        // X tile: 32x32 f32 = 256 float4 -> exactly one b128 load per thread
        {
            const int m = tid >> 3, k4 = (tid & 7) << 2;
            const float4 x4 = *(const float4*)&X[(size_t)(m0 + m) * DDIM + k0 + k4];
            bf16x4 t;
            t.v[0] = (bf16_t)x4.x; t.v[1] = (bf16_t)x4.y;
            t.v[2] = (bf16_t)x4.z; t.v[3] = (bf16_t)x4.w;
            *(bf16x4*)&Abf[m][k4] = t;
        }
        // W tile: 32x512 f32, b128 global reads, transposed bf16 LDS writes
#pragma unroll
        for (int j = 0; j < 16; ++j) {
            const int idx = tid + j * 256;          // [0, 4096) float4 units
            const int k = idx >> 7, n4 = (idx & 127) << 2;
            const float4 w4 = *(const float4*)&W[(size_t)(k0 + k) * PDIM + n4];
            BbfT[n4 + 0][k] = (bf16_t)w4.x;
            BbfT[n4 + 1][k] = (bf16_t)w4.y;
            BbfT[n4 + 2][k] = (bf16_t)w4.z;
            BbfT[n4 + 3][k] = (bf16_t)w4.w;
        }
        __syncthreads();
        v16bf a = load_frag_a(&Abf[16 * mt][0], 32, lane);
#pragma unroll
        for (int t = 0; t < 8; ++t) {
            v16bf b = load_frag_b(&BbfT[128 * ng + 16 * t][0], 32, lane);
            acc[t] = wmma_bf16(a, b, acc[t]);
        }
        __syncthreads();
    }

    // bias add + per-row stats, reduced across the 16 lanes of each half
    float sr[8], s2r[8];
#pragma unroll
    for (int r = 0; r < 8; ++r) { sr[r] = 0.f; s2r[r] = 0.f; }
#pragma unroll
    for (int t = 0; t < 8; ++t) {
        const int n = 128 * ng + 16 * t + l15;
        const float bq = bias[n];
#pragma unroll
        for (int r = 0; r < 8; ++r) {
            float x = acc[t][r] + bq;
            acc[t][r] = x;
            sr[r] += x; s2r[r] += x * x;
        }
    }
#pragma unroll
    for (int msk = 1; msk < 16; msk <<= 1) {
#pragma unroll
        for (int r = 0; r < 8; ++r) {
            sr[r]  += __shfl_xor(sr[r],  msk, 32);
            s2r[r] += __shfl_xor(s2r[r], msk, 32);
        }
    }
    if (l15 == 0) {
#pragma unroll
        for (int r = 0; r < 8; ++r) {
            const int row = 16 * mt + 8 * half + r;
            red[row][ng][0] = sr[r];
            red[row][ng][1] = s2r[r];
        }
    }
    __syncthreads();
    if (tid < 32) {
        float ss = 0.f, ss2 = 0.f;
#pragma unroll
        for (int j = 0; j < 4; ++j) { ss += red[tid][j][0]; ss2 += red[tid][j][1]; }
        const float mu = ss / (float)PDIM;
        const float var = ss2 / (float)PDIM - mu * mu;
        muArr[tid] = mu;
        rsArr[tid] = rsqrtf(var + LN_EPS);
    }
    __syncthreads();
#pragma unroll
    for (int t = 0; t < 8; ++t) {
        const int n = 128 * ng + 16 * t + l15;
        const float gn = g[n], bn = bvec[n];
#pragma unroll
        for (int r = 0; r < 8; ++r) {
            const int row = 16 * mt + 8 * half + r;
            const float x = (acc[t][r] - muArr[row]) * rsArr[row] * gn + bn;
            out[(size_t)(m0 + row) * PDIM + n] = (bf16_t)x;
        }
    }
}

// ======================= Kernel 2: sim = Qn @ Kn^T, masked =======================
// Block tile 64(M) x 128(N); wave grid 2x4, each wave 32x32 (2x2 WMMA tiles). K=512.
// Staging is a pure bf16 byte-copy -> use GLOBAL_LOAD_ASYNC_TO_LDS_B128.
__global__ __launch_bounds__(256) void sim_mask_kernel(
    const bf16_t* __restrict__ Qn, const bf16_t* __restrict__ Kn,
    const int* __restrict__ uids, float* __restrict__ simout)
{
    __shared__ bf16_t Qt[64][32];     // 4 KB
    __shared__ bf16_t Kt[128][32];    // 8 KB
    const int tid = threadIdx.x, wave = tid >> 5, lane = tid & 31;
    const int wm = wave & 1, wn = wave >> 1;
    const int half = (lane >> 4) & 1, l15 = lane & 15;
    const int m0 = blockIdx.y * 64, n0 = blockIdx.x * 128;

    const unsigned qtBase = (unsigned)(uintptr_t)&Qt[0][0];   // flat->LDS offset (low 32 bits)
    const unsigned ktBase = (unsigned)(uintptr_t)&Kt[0][0];

    v8f acc[2][2] = {};
    for (int k0 = 0; k0 < PDIM; k0 += 32) {
        {   // Qt: 4096 B -> one async b128 per thread
            const int r = tid >> 2, kc = (tid & 3) << 3;
            async_copy_b128(qtBase + tid * 16, &Qn[(size_t)(m0 + r) * PDIM + k0 + kc]);
        }
#pragma unroll
        for (int j = 0; j < 2; ++j) {   // Kt: 8192 B -> two async b128 per thread
            const int t2 = tid + j * 256;
            const int r = t2 >> 2, kc = (t2 & 3) << 3;
            async_copy_b128(ktBase + t2 * 16, &Kn[(size_t)(n0 + r) * PDIM + k0 + kc]);
        }
        wait_async_all();
        __syncthreads();
        v16bf a[2], b[2];
#pragma unroll
        for (int i = 0; i < 2; ++i) a[i] = load_frag_a(&Qt[32 * wm + 16 * i][0], 32, lane);
#pragma unroll
        for (int j = 0; j < 2; ++j) b[j] = load_frag_b(&Kt[32 * wn + 16 * j][0], 32, lane);
#pragma unroll
        for (int i = 0; i < 2; ++i)
#pragma unroll
            for (int j = 0; j < 2; ++j) acc[i][j] = wmma_bf16(a[i], b[j], acc[i][j]);
        __syncthreads();
    }
#pragma unroll
    for (int j = 0; j < 2; ++j) {
        const int n = n0 + 32 * wn + 16 * j + l15;
        const int un = uids[n];
#pragma unroll
        for (int i = 0; i < 2; ++i) {
#pragma unroll
            for (int r = 0; r < 8; ++r) {
                const int m = m0 + 32 * wm + 16 * i + 8 * half + r;
                float v = acc[i][j][r];
                if (m == n) v -= MASK_VAL;             // eye
                if (uids[m] == un) v -= MASK_VAL;      // same uid (diagonal masked twice, as ref)
                simout[(size_t)m * NROW + n] = v;
            }
        }
    }
}

// ======================= Kernel 3: row softmax in place =======================
__global__ __launch_bounds__(256) void softmax_kernel(float* __restrict__ sim)
{
    __shared__ float buf[NROW];
    __shared__ float red[256];
    const int tid = threadIdx.x;
    float* row = sim + (size_t)blockIdx.x * NROW;

    float mx = -3.4e38f;
    for (int i = tid; i < NROW; i += 256) { float v = row[i]; buf[i] = v; mx = fmaxf(mx, v); }
    red[tid] = mx; __syncthreads();
    for (int s = 128; s > 0; s >>= 1) { if (tid < s) red[tid] = fmaxf(red[tid], red[tid + s]); __syncthreads(); }
    mx = red[0]; __syncthreads();

    float sum = 0.f;
    for (int i = tid; i < NROW; i += 256) { float e = __expf(buf[i] - mx); buf[i] = e; sum += e; }
    red[tid] = sum; __syncthreads();
    for (int s = 128; s > 0; s >>= 1) { if (tid < s) red[tid] += red[tid + s]; __syncthreads(); }
    const float inv = 1.f / red[0];
    for (int i = tid; i < NROW; i += 256) row[i] = buf[i] * inv;
}

// ======================= Kernel 4: recon = sim_sm @ X =======================
// Block tile 64(M) x 128(N), K=4096 in 32-steps; f32 -> bf16 conversion at staging.
__global__ __launch_bounds__(256) void recon_kernel(
    const float* __restrict__ Pm, const float* __restrict__ X, float* __restrict__ out)
{
    __shared__ bf16_t At[64][32];
    __shared__ bf16_t BT[128][32];   // [n][k]
    const int tid = threadIdx.x, wave = tid >> 5, lane = tid & 31;
    const int wm = wave & 1, wn = wave >> 1;
    const int half = (lane >> 4) & 1, l15 = lane & 15;
    const int m0 = blockIdx.y * 64, n0 = blockIdx.x * 128;

    v8f acc[2][2] = {};
    for (int k0 = 0; k0 < NROW; k0 += 32) {
        // A tile: 64x32 f32 = 512 float4 -> two b128 loads per thread
#pragma unroll
        for (int j = 0; j < 2; ++j) {
            const int u = tid + j * 256;
            const int r = u >> 3, k4 = (u & 7) << 2;
            const float4 p4 = *(const float4*)&Pm[(size_t)(m0 + r) * NROW + k0 + k4];
            bf16x4 t;
            t.v[0] = (bf16_t)p4.x; t.v[1] = (bf16_t)p4.y;
            t.v[2] = (bf16_t)p4.z; t.v[3] = (bf16_t)p4.w;
            *(bf16x4*)&At[r][k4] = t;
        }
        // Prefetch next k-step of the A stream (64 MB total -> HBM bound phase)
        if (k0 + 32 < NROW) {
            const int r = tid >> 3, k4 = (tid & 7) << 2;
            __builtin_prefetch(&Pm[(size_t)(m0 + r) * NROW + k0 + 32 + k4], 0, 1);
        }
        // B tile: 32x128 f32, b128 reads over n, transposed bf16 writes
#pragma unroll
        for (int j = 0; j < 4; ++j) {
            const int u = tid + j * 256;           // [0, 1024) float4 units
            const int k = u >> 5, n4 = (u & 31) << 2;
            const float4 x4 = *(const float4*)&X[(size_t)(k0 + k) * DDIM + n0 + n4];
            BT[n4 + 0][k] = (bf16_t)x4.x;
            BT[n4 + 1][k] = (bf16_t)x4.y;
            BT[n4 + 2][k] = (bf16_t)x4.z;
            BT[n4 + 3][k] = (bf16_t)x4.w;
        }
        __syncthreads();
        v16bf a[2], b[2];
#pragma unroll
        for (int i = 0; i < 2; ++i) a[i] = load_frag_a(&At[32 * wm + 16 * i][0], 32, lane);
#pragma unroll
        for (int j = 0; j < 2; ++j) b[j] = load_frag_b(&BT[32 * wn + 16 * j][0], 32, lane);
#pragma unroll
        for (int i = 0; i < 2; ++i)
#pragma unroll
            for (int j = 0; j < 2; ++j) acc[i][j] = wmma_bf16(a[i], b[j], acc[i][j]);
        __syncthreads();
    }
#pragma unroll
    for (int i = 0; i < 2; ++i)
#pragma unroll
        for (int j = 0; j < 2; ++j)
#pragma unroll
            for (int r = 0; r < 8; ++r) {
                const int m = m0 + 32 * wm + 16 * i + 8 * half + r;
                const int n = n0 + 32 * wn + 16 * j + l15;
                out[(size_t)m * DDIM + n] = acc[i][j][r];
            }
}

extern "C" void kernel_launch(void* const* d_in, const int* in_sizes, int n_in,
                              void* d_out, int out_size, void* d_ws, size_t ws_size,
                              hipStream_t stream) {
    const float* sim1 = (const float*)d_in[0];
    const float* sim2 = (const float*)d_in[1];
    const float* xrec = (const float*)d_in[2];
    const float* Wq   = (const float*)d_in[3];
    const float* bq   = (const float*)d_in[4];
    const float* Wk   = (const float*)d_in[5];
    const float* bk   = (const float*)d_in[6];
    const float* lng  = (const float*)d_in[7];
    const float* lnb  = (const float*)d_in[8];
    const int*   uids = (const int*)d_in[12];

    float* recon = (float*)d_out;                       // [4096 x 2048]
    float* simsm = recon + (size_t)NROW * DDIM;         // [4096 x 4096]

    bf16_t* Qn = (bf16_t*)d_ws;                         // [4096 x 512] bf16
    bf16_t* Kn = Qn + (size_t)NROW * PDIM;              // [4096 x 512] bf16

    proj_ln_kernel<<<dim3(NROW / 32), 256, 0, stream>>>(sim1, Wq, bq, lng, lnb, Qn);
    proj_ln_kernel<<<dim3(NROW / 32), 256, 0, stream>>>(sim2, Wk, bk, lng, lnb, Kn);
    sim_mask_kernel<<<dim3(NROW / 128, NROW / 64), 256, 0, stream>>>(Qn, Kn, uids, simsm);
    softmax_kernel<<<dim3(NROW), 256, 0, stream>>>(simsm);
    recon_kernel<<<dim3(DDIM / 128, NROW / 64), 256, 0, stream>>>(simsm, xrec, recon);
}
